// Attention_25168508354908
// MI455X (gfx1250) — matrix operand order repaired and checked by reading, not compiled, not run on hardware
//
#include <hip/hip_runtime.h>

// ---------------------------------------------------------------------------
// Pooled-KV attention for MI455X (gfx1250, wave32).
// All matmuls: v_wmma_f32_16x16x32_bf16 (fp32 accumulate, bf16 operands).
// fp32->bf16 conversion hoisted out of all GEMM inner loops; x transposed once
// (staged via global_load_async_to_lds_b128) into a B-operand-ready layout, so
// the projection GEMMs are LDS-free.
// ---------------------------------------------------------------------------

typedef __attribute__((ext_vector_type(16))) __bf16 v16bf;
typedef __attribute__((ext_vector_type(8)))  __bf16 bfx8;
typedef __attribute__((ext_vector_type(8)))  float  v8f;

static __device__ __forceinline__ unsigned bfbits(float f) {
  union { float f; unsigned u; } a; a.f = f;
  return (a.u + 0x7FFFu + ((a.u >> 16) & 1u)) >> 16;  // RNE truncation
}
static __device__ __forceinline__ __bf16 f2bf(float f) {
  union { unsigned short s; __bf16 b; } o; o.s = (unsigned short)bfbits(f);
  return o.b;
}
static __device__ __forceinline__ unsigned pack2bf(float lo, float hi) {
  return bfbits(lo) | (bfbits(hi) << 16);
}

static __device__ __forceinline__ v8f wmma_bf16(v16bf a, v16bf b, v8f c) {
  return __builtin_amdgcn_wmma_f32_16x16x32_bf16(false, a, false, b, (short)0, c,
                                                 false, false);
}

// A-tile (16x32) from a bf16 row-major matrix.  Layout: lanes 0-15 row
// M=lane hold K {0..15}; lanes 16-31 row M=lane-16 hold K {16..31}.
static __device__ __forceinline__ v16bf load_a_bf(const __bf16* __restrict__ W,
                                                  int ld, int row, int kbase,
                                                  int hi) {
  const __bf16* p = W + (size_t)row * ld + kbase + hi * 16;
  bfx8 lo = *(const bfx8*)p;
  bfx8 hh = *(const bfx8*)(p + 8);
  return __builtin_shufflevector(lo, hh, 0, 1, 2, 3, 4, 5, 6, 7,
                                 8, 9, 10, 11, 12, 13, 14, 15);
}

static __device__ __forceinline__ void wait_async0() {
#if __has_builtin(__builtin_amdgcn_s_wait_asynccnt)
  __builtin_amdgcn_s_wait_asynccnt(0);
#else
  asm volatile("s_wait_asynccnt 0x0" ::: "memory");
#endif
}

// ---------------------------------------------------------------------------
// 0) Wqkv (1536x512) + Wout (512x512) -> bf16 wbf (packed u32 stores).
// ---------------------------------------------------------------------------
__global__ void __launch_bounds__(256)
convert_w_kernel(const float* __restrict__ Wqkv, const float* __restrict__ Wout,
                 __bf16* __restrict__ wbf) {
  unsigned* dst = (unsigned*)wbf;
#pragma unroll
  for (int k = 0; k < 4; ++k) {
    unsigned p = (blockIdx.x * 256 + threadIdx.x) * 4 + k;
    unsigned e = p * 2;
    float a, b;
    if (e < 786432u) { a = Wqkv[e]; b = Wqkv[e + 1]; }
    else             { a = Wout[e - 786432u]; b = Wout[e - 786432u + 1]; }
    dst[p] = pack2bf(a, b);
  }
}

// ---------------------------------------------------------------------------
// 1) pool x -> xpT bf16 [B, n=320, c=512]  (B-operand-ready for the kv GEMM).
//    Pooling is linear so it commutes with the 1x1 conv.
// ---------------------------------------------------------------------------
__global__ void __launch_bounds__(128)
pool_kernel(const float* __restrict__ x, __bf16* __restrict__ xpT) {
  const int b = blockIdx.x >> 9, c = blockIdx.x & 511;
  __shared__ float plane[1024];
  const float* src = x + (size_t)blockIdx.x * 1024;
  for (int i = threadIdx.x; i < 1024; i += 128) plane[i] = src[i];
  __syncthreads();
  for (int i = threadIdx.x; i < 320; i += 128) {
    float v = 0.f;
    if (i < 32) {                       // mean over width
      for (int w = 0; w < 32; ++w) v += plane[i * 32 + w];
      v *= (1.f / 32.f);
    } else if (i < 64) {                // mean over height
      int w = i - 32;
      for (int hh = 0; hh < 32; ++hh) v += plane[hh * 32 + w];
      v *= (1.f / 32.f);
    } else {                            // 2x2 average pool
      int p = i - 64, ph = p >> 4, pw = p & 15;
      int base = (ph * 2) * 32 + pw * 2;
      v = 0.25f * (plane[base] + plane[base + 1] + plane[base + 32] + plane[base + 33]);
    }
    xpT[((size_t)b * 320 + i) * 512 + c] = f2bf(v);
  }
}

// ---------------------------------------------------------------------------
// 2) x [b][c][s] fp32 -> xT bf16 [b][s=1024][c=512].  64x64 tiles staged into
//    LDS with GLOBAL_LOAD_ASYNC_TO_LDS_B128 (ASYNCcnt/s_wait_asynccnt), then
//    transposed reads + packed bf16 coalesced stores.
// ---------------------------------------------------------------------------
__global__ void __launch_bounds__(256)
transpose_x_kernel(const float* __restrict__ x, __bf16* __restrict__ xT) {
  const int b = blockIdx.z;
  const int s0 = blockIdx.x * 64, c0 = blockIdx.y * 64;
  const int tid = threadIdx.x;
  __shared__ __attribute__((aligned(16))) float tile[64][68];  // [c][s], padded

#pragma unroll
  for (int k = 0; k < 4; ++k) {
    int id = tid + 256 * k;
    int cc = id >> 4, c16 = id & 15;             // row, 16B chunk within row
    const float* gp = x + ((size_t)b * 512 + c0 + cc) * 1024 + s0 + c16 * 4;
    unsigned laddr = (unsigned)(size_t)&tile[cc][c16 * 4];  // LDS byte offset
    asm volatile("global_load_async_to_lds_b128 %0, %1, off"
                 :: "v"(laddr), "v"(gp) : "memory");
  }
  wait_async0();
  __syncthreads();

#pragma unroll
  for (int k = 0; k < 8; ++k) {
    int p = tid + 256 * k;
    int c2 = p & 31, ss = p >> 5;                // c-pair, s row
    float a = tile[2 * c2][ss];
    float bv = tile[2 * c2 + 1][ss];
    unsigned* o = (unsigned*)(xT + ((size_t)b * 1024 + s0 + ss) * 512 + c0 + 2 * c2);
    *o = pack2bf(a, bv);
  }
}

// ---------------------------------------------------------------------------
// 3) q = Wq @ x -> q bf16 [B,h,s,d].  LDS-free; wave = 16(M) x 128(N).
// ---------------------------------------------------------------------------
__global__ void __launch_bounds__(256)
gemm_q_kernel(const __bf16* __restrict__ wbf, const __bf16* __restrict__ xT,
              __bf16* __restrict__ qbf) {
  const int b = blockIdx.z;
  const int tid = threadIdx.x;
  const int wave = tid >> 5, lane = tid & 31, l16 = lane & 15, hi = lane >> 4;
  const int o_tile = blockIdx.y * 128 + wave * 16;
  const int s_base = blockIdx.x * 128;
  v8f acc[8] = {};
  for (int kk = 0; kk < 16; ++kk) {
    v16bf a = load_a_bf(wbf, 512, o_tile + l16, kk * 32, hi);
    if (kk < 15)
      __builtin_prefetch(wbf + (size_t)(o_tile + l16) * 512 + (kk + 1) * 32, 0, 3);
#pragma unroll
    for (int nt = 0; nt < 8; ++nt) {
      int s = s_base + nt * 16 + l16;
      v16bf bv = *(const v16bf*)(xT + ((size_t)b * 1024 + s) * 512 + kk * 32 + hi * 16);
      acc[nt] = wmma_bf16(a, bv, acc[nt]);
    }
  }
#pragma unroll
  for (int nt = 0; nt < 8; ++nt)
#pragma unroll
    for (int r = 0; r < 8; ++r) {
      int o = o_tile + r + hi * 8;
      int s = s_base + nt * 16 + l16;
      int h = o >> 6, d = o & 63;
      qbf[(((size_t)b * 8 + h) * 1024 + s) * 64 + d] = f2bf(acc[nt][r]);
    }
}

// ---------------------------------------------------------------------------
// 4) kv = sigmoid(Wkv @ xp): k bf16 [B,h,n,d], v bf16 transposed [B,h,d,n].
// ---------------------------------------------------------------------------
__global__ void __launch_bounds__(256)
gemm_kv_kernel(const __bf16* __restrict__ wbf, const __bf16* __restrict__ xpT,
               __bf16* __restrict__ kbf, __bf16* __restrict__ vbf) {
  const int b = blockIdx.z;
  const int tid = threadIdx.x;
  const int wave = tid >> 5, lane = tid & 31, l16 = lane & 15, hi = lane >> 4;
  const int o_tile = blockIdx.y * 128 + wave * 16;   // rows of [Wk;Wv] 0..1023
  const int n_base = blockIdx.x * 64;
  const __bf16* Wkv = wbf + (size_t)512 * 512;
  v8f acc[4] = {};
  for (int kk = 0; kk < 16; ++kk) {
    v16bf a = load_a_bf(Wkv, 512, o_tile + l16, kk * 32, hi);
#pragma unroll
    for (int nt = 0; nt < 4; ++nt) {
      int n = n_base + nt * 16 + l16;
      v16bf bv = *(const v16bf*)(xpT + ((size_t)b * 320 + n) * 512 + kk * 32 + hi * 16);
      acc[nt] = wmma_bf16(a, bv, acc[nt]);
    }
  }
#pragma unroll
  for (int nt = 0; nt < 4; ++nt)
#pragma unroll
    for (int r = 0; r < 8; ++r) {
      int o = o_tile + r + hi * 8;
      int n = n_base + nt * 16 + l16;
      float sg = 1.f / (1.f + __expf(-acc[nt][r]));  // gate
      if (o < 512) {
        int hh = o >> 6, d = o & 63;
        kbf[(((size_t)b * 8 + hh) * 320 + n) * 64 + d] = f2bf(sg);
      } else {
        int oo = o - 512, hh = oo >> 6, d = oo & 63;
        vbf[(((size_t)b * 8 + hh) * 64 + d) * 320 + n] = f2bf(sg);
      }
    }
}

// ---------------------------------------------------------------------------
// 5) attention: per (b,h,64-row s-tile); 4 waves x 16 rows; 80 WMMA/wave.
// ---------------------------------------------------------------------------
__global__ void __launch_bounds__(128)
attn_kernel(const __bf16* __restrict__ qbf, const __bf16* __restrict__ kbf,
            const __bf16* __restrict__ vbf, const float* __restrict__ pos,
            __bf16* __restrict__ obf) {
  const int b = blockIdx.z, h = blockIdx.y;
  const int tid = threadIdx.x;
  const int wave = tid >> 5, lane = tid & 31, l16 = lane & 15, hi = lane >> 4;
  const int s0 = blockIdx.x * 64 + wave * 16;
  const size_t qb = (((size_t)b * 8 + h) * 1024) * 64;
  const size_t kb = (((size_t)b * 8 + h) * 320) * 64;
  const size_t vb = (((size_t)b * 8 + h) * 64) * 320;

  __shared__ __attribute__((aligned(32))) __bf16 p_lds[4][16][328];

  v8f acc[20] = {};
#pragma unroll
  for (int kk = 0; kk < 2; ++kk) {
    v16bf a = load_a_bf(qbf + qb, 64, s0 + l16, kk * 32, hi);
#pragma unroll
    for (int t = 0; t < 20; ++t) {
      const __bf16* kp = kbf + kb + (size_t)(t * 16 + l16) * 64 + kk * 32 + hi * 16;
      v16bf bv = *(const v16bf*)kp;
      acc[t] = wmma_bf16(a, bv, acc[t]);
    }
  }

  const float scale = 0.125f;  // d^-0.5
#pragma unroll
  for (int r = 0; r < 8; ++r) {
    const int srow = s0 + r + hi * 8;
    float m = -3.0e38f;
#pragma unroll
    for (int t = 0; t < 20; ++t) {
      float v = acc[t][r] * scale + pos[(size_t)srow * 320 + t * 16 + l16];
      acc[t][r] = v;
      m = fmaxf(m, v);
    }
#pragma unroll
    for (int off = 1; off < 16; off <<= 1) m = fmaxf(m, __shfl_xor(m, off, 32));
    float sum = 0.f;
#pragma unroll
    for (int t = 0; t < 20; ++t) {
      float e = __expf(acc[t][r] - m);
      acc[t][r] = e;
      sum += e;
    }
#pragma unroll
    for (int off = 1; off < 16; off <<= 1) sum += __shfl_xor(sum, off, 32);
    float inv = 1.f / sum;
#pragma unroll
    for (int t = 0; t < 20; ++t)
      p_lds[wave][r + hi * 8][t * 16 + l16] = f2bf(acc[t][r] * inv);
  }
  __syncthreads();

  v8f oacc[4] = {};
#pragma unroll
  for (int kk = 0; kk < 10; ++kk) {
    const __bf16* pp = &p_lds[wave][l16][kk * 32 + hi * 16];
    bfx8 lo = *(const bfx8*)pp;
    bfx8 hh = *(const bfx8*)(pp + 8);
    v16bf a = __builtin_shufflevector(lo, hh, 0, 1, 2, 3, 4, 5, 6, 7,
                                      8, 9, 10, 11, 12, 13, 14, 15);
#pragma unroll
    for (int dt = 0; dt < 4; ++dt) {
      const __bf16* vp = vbf + vb + (size_t)(dt * 16 + l16) * 320 + kk * 32 + hi * 16;
      v16bf bv = *(const v16bf*)vp;
      oacc[dt] = wmma_bf16(a, bv, oacc[dt]);
    }
  }
#pragma unroll
  for (int dt = 0; dt < 4; ++dt)
#pragma unroll
    for (int r = 0; r < 8; ++r) {
      int s = s0 + r + hi * 8;
      int c = h * 64 + dt * 16 + l16;
      obf[((size_t)b * 1024 + s) * 512 + c] = f2bf(oacc[dt][r]);
    }
}

// ---------------------------------------------------------------------------
// 6) out = Wout @ O + bout -> fp32 [B,512,32,32].  LDS-free.
// ---------------------------------------------------------------------------
__global__ void __launch_bounds__(256)
gemm_out_kernel(const __bf16* __restrict__ wbf, const __bf16* __restrict__ obf,
                const float* __restrict__ bout, float* __restrict__ out) {
  const int b = blockIdx.z;
  const int tid = threadIdx.x;
  const int wave = tid >> 5, lane = tid & 31, l16 = lane & 15, hi = lane >> 4;
  const int o_tile = blockIdx.y * 128 + wave * 16;
  const int s_base = blockIdx.x * 128;
  const __bf16* Wo = wbf + (size_t)1536 * 512;
  v8f acc[8] = {};
  for (int kk = 0; kk < 16; ++kk) {
    v16bf a = load_a_bf(Wo, 512, o_tile + l16, kk * 32, hi);
    if (kk < 15)
      __builtin_prefetch(Wo + (size_t)(o_tile + l16) * 512 + (kk + 1) * 32, 0, 3);
#pragma unroll
    for (int nt = 0; nt < 8; ++nt) {
      int s = s_base + nt * 16 + l16;
      v16bf bv = *(const v16bf*)(obf + ((size_t)b * 1024 + s) * 512 + kk * 32 + hi * 16);
      acc[nt] = wmma_bf16(a, bv, acc[nt]);
    }
  }
#pragma unroll
  for (int nt = 0; nt < 8; ++nt)
#pragma unroll
    for (int r = 0; r < 8; ++r) {
      int o = o_tile + r + hi * 8;
      int s = s_base + nt * 16 + l16;
      out[((size_t)b * 512 + o) * 1024 + s] = acc[nt][r] + bout[o];
    }
}

// ---------------------------------------------------------------------------
// Workspace (bytes), total 128 MB:
//   xT@0  qbf@33554432  kbf@67108864  vbf@77594624  obf@88080384
//   xpT@121634816  wbf@132120576
// ---------------------------------------------------------------------------
extern "C" void kernel_launch(void* const* d_in, const int* in_sizes, int n_in,
                              void* d_out, int out_size, void* d_ws, size_t ws_size,
                              hipStream_t stream) {
  (void)in_sizes; (void)n_in; (void)out_size; (void)ws_size;
  const float* x    = (const float*)d_in[0];
  const float* Wqkv = (const float*)d_in[1];
  const float* Wout = (const float*)d_in[2];
  const float* bout = (const float*)d_in[3];
  const float* pos  = (const float*)d_in[4];
  float* out = (float*)d_out;

  char* ws = (char*)d_ws;
  __bf16* xT  = (__bf16*)(ws);
  __bf16* qbf = (__bf16*)(ws + 33554432u);
  __bf16* kbf = (__bf16*)(ws + 67108864u);
  __bf16* vbf = (__bf16*)(ws + 77594624u);
  __bf16* obf = (__bf16*)(ws + 88080384u);
  __bf16* xpT = (__bf16*)(ws + 121634816u);
  __bf16* wbf = (__bf16*)(ws + 132120576u);

  convert_w_kernel  <<<dim3(640),       dim3(256), 0, stream>>>(Wqkv, Wout, wbf);
  pool_kernel       <<<dim3(32 * 512),  dim3(128), 0, stream>>>(x, xpT);
  transpose_x_kernel<<<dim3(16, 8, 32), dim3(256), 0, stream>>>(x, xT);
  gemm_q_kernel     <<<dim3(8, 4, 32),  dim3(256), 0, stream>>>(wbf, xT, qbf);
  gemm_kv_kernel    <<<dim3(5, 8, 32),  dim3(256), 0, stream>>>(wbf, xpT, kbf, vbf);
  attn_kernel       <<<dim3(16, 8, 32), dim3(128), 0, stream>>>(qbf, kbf, vbf, pos, obf);
  gemm_out_kernel   <<<dim3(8, 4, 32),  dim3(256), 0, stream>>>(wbf, obf, bout, out);
}
